// GraphMultiHeadAttention_31568009626228
// MI455X (gfx1250) — compile-verified, matrix operand order checked by reading, not run on hardware
//
#include <hip/hip_runtime.h>
#include <hip/hip_bf16.h>

// ---------------------------------------------------------------------------
// GraphMultiHeadAttention for MI455X (gfx1250, wave32, WMMA bf16 16x16x32)
//   B=8, N=1024, HID=1024, H=16, D=64, SCALE = 0.125
// Pipeline:
//   1) cast activations fp32->bf16; transpose+cast weights to [Nc x K] bf16
//   2) WMMA GEMMs: Q=(Xq Wq+bq)*SCALE -> [B,H,N,D]; K -> [B,H,N,D]; V -> [B,H,D,N]
//   3) flash-attention; the 512MB attn_bias stream is double-buffered through
//      LDS with gfx1250 async global->LDS b128 loads (ASYNCcnt)
//   4) WMMA GEMM: out = ctx Wo + bo (fp32)
// ---------------------------------------------------------------------------

typedef __bf16 bf16_t;
typedef __attribute__((ext_vector_type(8)))  __bf16 v8bf;
typedef __attribute__((ext_vector_type(16))) __bf16 v16bf;
typedef __attribute__((ext_vector_type(8)))  float  v8f;

#define BB   8
#define NN   1024
#define HID  1024
#define HH   16
#define DD   64
#define TOK  (BB * NN)        // 8192
#define SCALE 0.125f

// ---- CDNA5 async global->LDS (ASYNCcnt) -----------------------------------
// One b128 per lane: 32 lanes move 512B per instruction into LDS.
__device__ __forceinline__ void async_load_b128(const void* gsrc, void* lds_dst) {
  // LDS byte offset = low 32 bits of the generic shared-aperture address.
  unsigned lds_off = (unsigned)(unsigned long long)lds_dst;
  unsigned long long gaddr = (unsigned long long)gsrc;
  asm volatile("global_load_async_to_lds_b128 %0, %1, off"
               :: "v"(lds_off), "v"(gaddr) : "memory");
}
#define WAIT_ASYNC(n) asm volatile("s_wait_asynccnt " #n ::: "memory")

// Fetch a 16x32 fp32 bias tile (rows stride NN) into LDS, 4 async b128 ops.
__device__ __forceinline__ void async_bias_tile(const float* gsrc, float* ldst) {
  const int lane = threadIdx.x & 31;
  const int rr = lane >> 3;        // 4 rows per instruction
  const int cc = (lane & 7) << 2;  // 8 lanes x 4 floats = 32 cols
#pragma unroll
  for (int it = 0; it < 4; ++it) {
    const int m = it * 4 + rr;
    async_load_b128(gsrc + (size_t)m * NN + cc, ldst + m * 32 + cc);
  }
}

// ---- WMMA fragment loader -------------------------------------------------
// Row-major source: element(row, k). A-frag: row = M (lane&15); B-frag: row = N col.
// ISA 7.12.2 16-bit layout: lanes 0-15 hold K {0..7,16..23}, lanes 16-31 hold
// K {8..15,24..31}; each VGPR packs two consecutive K. -> two 16B loads/lane.
__device__ __forceinline__ v16bf load_frag(const bf16_t* base, int ld) {
  const int lane = threadIdx.x & 31;
  const bf16_t* p = base + (size_t)(lane & 15) * ld + ((lane >> 4) << 3);
  v8bf lo = *(const v8bf*)(p);
  v8bf hi = *(const v8bf*)(p + 16);
  v16bf r;
#pragma unroll
  for (int i = 0; i < 8; ++i) { r[i] = lo[i]; r[i + 8] = hi[i]; }
  return r;
}

__device__ __forceinline__ v8f wmma_bf16(v16bf a, v16bf b, v8f c) {
  return __builtin_amdgcn_wmma_f32_16x16x32_bf16(false, a, false, b,
                                                 (short)0, c, false, false);
}

// reductions across a 16-lane group (rows of a C-tile live on 16 lanes)
__device__ __forceinline__ float rowmax16(float v) {
#pragma unroll
  for (int m = 8; m >= 1; m >>= 1) v = fmaxf(v, __shfl_xor(v, m, 32));
  return v;
}
__device__ __forceinline__ float rowsum16(float v) {
#pragma unroll
  for (int m = 8; m >= 1; m >>= 1) v += __shfl_xor(v, m, 32);
  return v;
}

// ---- 1) casts -------------------------------------------------------------
__global__ __launch_bounds__(256) void cast_bf16_kernel(const float* __restrict__ x,
                                                        bf16_t* __restrict__ y, int n) {
  for (int i = blockIdx.x * blockDim.x + threadIdx.x; i < n;
       i += gridDim.x * blockDim.x)
    y[i] = (bf16_t)x[i];
}

// Wt[n][k] = (bf16)W[k][n]   (1024x1024), LDS 32x32 tile transpose
__global__ __launch_bounds__(256) void transpose_cast_kernel(const float* __restrict__ W,
                                                             bf16_t* __restrict__ Wt) {
  __shared__ float tile[32][33];
  const int bx = blockIdx.x, by = blockIdx.y;   // bx: n-tile, by: k-tile
  const int tx = threadIdx.x, ty = threadIdx.y; // 32 x 8
#pragma unroll
  for (int i = ty; i < 32; i += 8)
    tile[i][tx] = W[(size_t)(by * 32 + i) * HID + bx * 32 + tx];
  __syncthreads();
#pragma unroll
  for (int i = ty; i < 32; i += 8)
    Wt[(size_t)(bx * 32 + i) * HID + by * 32 + tx] = (bf16_t)tile[tx][i];
}

// ---- 2/4) generic WMMA GEMM: C[8192 x 1024] = A[8192 x 1024] @ Wt^T + bias -
// A row-major bf16 (ld=1024); Wt is [1024 cols x 1024 k] row-major bf16.
// mode 0: fp32 out row-major; mode 1: bf16 out [B,H,N,D] * scale;
// mode 2: bf16 out [B,H,D,N].
__global__ __launch_bounds__(256)
void gemm_wmma_kernel(const bf16_t* __restrict__ A, const bf16_t* __restrict__ Wt,
                      const float* __restrict__ bias, float* __restrict__ outF,
                      bf16_t* __restrict__ outB, int mode, float scale) {
  const int w    = threadIdx.x >> 5;           // 8 waves
  const int lane = threadIdx.x & 31;
  const int row0 = blockIdx.x * 128 + (w & 3) * 32;
  const int col0 = blockIdx.y * 64 + (w >> 2) * 32;

  v8f acc[2][2] = {};
  for (int kk = 0; kk < HID; kk += 32) {
    v16bf a0 = load_frag(A + (size_t)row0 * HID + kk, HID);
    v16bf a1 = load_frag(A + (size_t)(row0 + 16) * HID + kk, HID);
    v16bf b0 = load_frag(Wt + (size_t)col0 * HID + kk, HID);
    v16bf b1 = load_frag(Wt + (size_t)(col0 + 16) * HID + kk, HID);
    acc[0][0] = wmma_bf16(a0, b0, acc[0][0]);
    acc[0][1] = wmma_bf16(a0, b1, acc[0][1]);
    acc[1][0] = wmma_bf16(a1, b0, acc[1][0]);
    acc[1][1] = wmma_bf16(a1, b1, acc[1][1]);
  }

  const int lh = lane >> 4, ln = lane & 15;
#pragma unroll
  for (int i = 0; i < 2; ++i)
#pragma unroll
    for (int j = 0; j < 2; ++j)
#pragma unroll
      for (int r = 0; r < 8; ++r) {
        const int m = row0 + 16 * i + r + 8 * lh;      // token row
        const int c = col0 + 16 * j + ln;              // feature col
        float v = acc[i][j][r] + bias[c];
        if (mode == 0) {
          outF[(size_t)m * HID + c] = v;
        } else {
          const int b = m >> 10, nt = m & 1023;
          const int h = c >> 6, d = c & 63;
          if (mode == 1)   // [B,H,N,D]
            outB[(((size_t)(b * HH + h) << 10) + nt) * DD + d] = (bf16_t)(v * scale);
          else             // [B,H,D,N]
            outB[(((size_t)(b * HH + h) * DD + d) << 10) + nt] = (bf16_t)v;
        }
      }
}

// ---- 3) flash attention ---------------------------------------------------
// One wave owns a 16-query block of one (b,h). Q pre-scaled. 32 keys/step.
// attn_bias tiles are double-buffered through LDS with async b128 loads.
__global__ __launch_bounds__(256)
void attn_wmma_kernel(const bf16_t* __restrict__ Q,   // [B,H,N,D]
                      const bf16_t* __restrict__ K,   // [B,H,N,D]
                      const bf16_t* __restrict__ Vt,  // [B,H,D,N]
                      const float* __restrict__ abias,// [B,H,N,N]
                      bf16_t* __restrict__ ctx) {     // [B,N,HID]
  __shared__ bf16_t plds[8 * 16 * 32];                // per-wave P tile (8KB)
  __shared__ float  blds[8 * 2 * 16 * 32];            // bias double-buffer (32KB)
  const int w    = threadIdx.x >> 5;
  const int lane = threadIdx.x & 31;
  const int lh = lane >> 4, ln = lane & 15;

  const int gw = blockIdx.x * 8 + w;                  // 0..8191
  const int b  = gw >> 10;                            // / (H * N/16)
  const int h  = (gw >> 6) & 15;
  const int q0 = (gw & 63) << 4;

  const size_t bh = (size_t)(b * HH + h);
  const bf16_t* Qf = Q + (bh * NN + q0) * DD;
  const bf16_t* Kf = K + bh * NN * DD;
  const bf16_t* Vf = Vt + bh * DD * NN;
  const float*  Bf = abias + ((bh * NN + q0) * NN);

  const v16bf qa0 = load_frag(Qf + 0, DD);            // features 0..31
  const v16bf qa1 = load_frag(Qf + 32, DD);           // features 32..63

  float m_i[8], l_i[8];
#pragma unroll
  for (int r = 0; r < 8; ++r) { m_i[r] = -3.0e38f; l_i[r] = 0.f; }
  v8f cacc[4] = {};                                   // 16 x 64 context

  bf16_t* pw = plds + w * 512;
  float* bb[2] = { blds + w * 1024, blds + w * 1024 + 512 };

  // prime the bias pipeline: tile for keys [0,32)
  async_bias_tile(Bf, bb[0]);
  int cur = 0;

  for (int j0 = 0; j0 < NN; j0 += 32) {
    // prefetch next bias tile while we compute on the current one
    if (j0 + 32 < NN) {
      async_bias_tile(Bf + j0 + 32, bb[cur ^ 1]);
      WAIT_ASYNC(4);          // in-order: current tile's 4 ops are complete
    } else {
      WAIT_ASYNC(0);
    }
    const float* bt = bb[cur];

    // S = Q @ K^T  (two 16x16 key sub-tiles)
    const bf16_t* k0b = Kf + (size_t)j0 * DD;
    const bf16_t* k1b = Kf + (size_t)(j0 + 16) * DD;
    v8f s0 = {}, s1 = {};
    s0 = wmma_bf16(qa0, load_frag(k0b + 0, DD), s0);
    s0 = wmma_bf16(qa1, load_frag(k0b + 32, DD), s0);
    s1 = wmma_bf16(qa0, load_frag(k1b + 0, DD), s1);
    s1 = wmma_bf16(qa1, load_frag(k1b + 32, DD), s1);

    // + bias (from LDS), online softmax (C layout: row r + 8*lh, col ln)
#pragma unroll
    for (int r = 0; r < 8; ++r) {
      const int m = r + 8 * lh;
      s0[r] += bt[m * 32 + ln];
      s1[r] += bt[m * 32 + 16 + ln];
      float mx = rowmax16(fmaxf(s0[r], s1[r]));
      float mn = fmaxf(m_i[r], mx);
      float al = __expf(m_i[r] - mn);
      m_i[r] = mn;
      float e0 = __expf(s0[r] - mn);
      float e1 = __expf(s1[r] - mn);
      l_i[r] = l_i[r] * al + rowsum16(e0 + e1);
      s0[r] = e0; s1[r] = e1;
      cacc[0][r] *= al; cacc[1][r] *= al; cacc[2][r] *= al; cacc[3][r] *= al;
    }

    // C-layout -> A-layout transpose of P through per-wave LDS (16x32 bf16)
#pragma unroll
    for (int r = 0; r < 8; ++r) {
      const int m = r + 8 * lh;
      pw[m * 32 + ln]      = (bf16_t)s0[r];
      pw[m * 32 + 16 + ln] = (bf16_t)s1[r];
    }
    const v16bf pf = load_frag(pw, 32);

    // ctx += P @ V  (V^T layout: row = d, keys contiguous)
#pragma unroll
    for (int dt = 0; dt < 4; ++dt) {
      v16bf vf = load_frag(Vf + (size_t)(dt * 16) * NN + j0, NN);
      cacc[dt] = wmma_bf16(pf, vf, cacc[dt]);
    }
    cur ^= 1;
  }

  // normalize and write ctx bf16 [B, N, H*D]
#pragma unroll
  for (int r = 0; r < 8; ++r) {
    const float inv = 1.0f / l_i[r];
    const int m = r + 8 * lh;
    const size_t rowb = ((size_t)b * NN + q0 + m) * HID + h * DD;
#pragma unroll
    for (int dt = 0; dt < 4; ++dt)
      ctx[rowb + dt * 16 + ln] = (bf16_t)(cacc[dt][r] * inv);
  }
}

// ---------------------------------------------------------------------------
extern "C" void kernel_launch(void* const* d_in, const int* in_sizes, int n_in,
                              void* d_out, int out_size, void* d_ws, size_t ws_size,
                              hipStream_t stream) {
  const float* Xq = (const float*)d_in[0];
  const float* Xk = (const float*)d_in[1];
  const float* Xv = (const float*)d_in[2];
  const float* AB = (const float*)d_in[3];
  const float* Wq = (const float*)d_in[4];  const float* bq = (const float*)d_in[5];
  const float* Wk = (const float*)d_in[6];  const float* bk = (const float*)d_in[7];
  const float* Wv = (const float*)d_in[8];  const float* bv = (const float*)d_in[9];
  const float* Wo = (const float*)d_in[10]; const float* bo = (const float*)d_in[11];
  float* out = (float*)d_out;

  size_t off = 0;
  auto carve = [&](size_t elems) -> bf16_t* {
    bf16_t* p = (bf16_t*)((char*)d_ws + off);
    off += (elems * sizeof(bf16_t) + 255) & ~(size_t)255;
    return p;
  };
  bf16_t* Xqb = carve((size_t)TOK * HID);
  bf16_t* Xkb = carve((size_t)TOK * HID);
  bf16_t* Xvb = carve((size_t)TOK * HID);
  bf16_t* WqT = carve((size_t)HID * HID);
  bf16_t* WkT = carve((size_t)HID * HID);
  bf16_t* WvT = carve((size_t)HID * HID);
  bf16_t* WoT = carve((size_t)HID * HID);
  bf16_t* Qb  = carve((size_t)TOK * HID);   // [B,H,N,D]
  bf16_t* Kb  = carve((size_t)TOK * HID);   // [B,H,N,D]
  bf16_t* Vtb = carve((size_t)TOK * HID);   // [B,H,D,N]
  bf16_t* Ctx = carve((size_t)TOK * HID);   // [B,N,HID]

  const int nAct = TOK * HID;
  cast_bf16_kernel<<<2048, 256, 0, stream>>>(Xq, Xqb, nAct);
  cast_bf16_kernel<<<2048, 256, 0, stream>>>(Xk, Xkb, nAct);
  cast_bf16_kernel<<<2048, 256, 0, stream>>>(Xv, Xvb, nAct);

  dim3 tb(32, 8), tg(32, 32);
  transpose_cast_kernel<<<tg, tb, 0, stream>>>(Wq, WqT);
  transpose_cast_kernel<<<tg, tb, 0, stream>>>(Wk, WkT);
  transpose_cast_kernel<<<tg, tb, 0, stream>>>(Wv, WvT);
  transpose_cast_kernel<<<tg, tb, 0, stream>>>(Wo, WoT);

  dim3 gg(TOK / 128, HID / 64);
  gemm_wmma_kernel<<<gg, 256, 0, stream>>>(Xqb, WqT, bq, nullptr, Qb, 1, SCALE);
  gemm_wmma_kernel<<<gg, 256, 0, stream>>>(Xkb, WkT, bk, nullptr, Kb, 1, 1.0f);
  gemm_wmma_kernel<<<gg, 256, 0, stream>>>(Xvb, WvT, bv, nullptr, Vtb, 2, 1.0f);

  attn_wmma_kernel<<<(BB * HH * (NN / 16)) / 8, 256, 0, stream>>>(Qb, Kb, Vtb, AB, Ctx);

  gemm_wmma_kernel<<<gg, 256, 0, stream>>>(Ctx, WoT, bo, out, nullptr, 0, 1.0f);
}
